// Model_45569603010903
// MI455X (gfx1250) — matrix operand order repaired and checked by reading, not compiled, not run on hardware
//
#include <hip/hip_runtime.h>
#include <hip/hip_bf16.h>

// ---------------------------------------------------------------------------
// MI455X (gfx1250) fused implementation.
//   Z: zero BN stats        A: conv+BN1 stats, store x_pre as f16
//   C: per-sample fused BN1+corr+graphconv+fc via v_wmma_f32_16x16x32_f16
//   E: BN2+sigmoid+classifier
// ---------------------------------------------------------------------------

typedef _Float16 v16h __attribute__((ext_vector_type(16)));
typedef _Float16 v8h  __attribute__((ext_vector_type(8)));
typedef _Float16 v2h  __attribute__((ext_vector_type(2)));
typedef float    v8f  __attribute__((ext_vector_type(8)));

#define BATCH 4096
#define CH    32
#define TT    512
#define NOUT  16
#define NCLS  4
#define EPS   1e-5f

// ---- WMMA helpers ----------------------------------------------------------
__device__ __forceinline__ v8f wmma16(v16h a, v16h b, v8f c) {
  // D = A(16x32 f16) * B(32x16 f16) + C(16x16 f32)
  return __builtin_amdgcn_wmma_f32_16x16x32_f16(
      /*neg_a=*/false, a, /*neg_b=*/false, b,
      /*c_mod=*/(short)0, c, /*reuse_a=*/false, /*reuse_b=*/false);
}

// A-fragment: row = m0 + (lane&15); lanes<16 hold K = k0+{0..7,16..23},
// lanes>=16 hold K = k0+{8..15,24..31}; 2 consecutive K per VGPR.
__device__ __forceinline__ v16h load_frag_A(const _Float16* base, int pitch,
                                            int m0, int k0, int lane) {
  int row = m0 + (lane & 15);
  int sel = (lane >> 4) & 1;
  const _Float16* p = base + row * pitch + k0 + sel * 8;
  v8h lo = *(const v8h*)(p);
  v8h hi = *(const v8h*)(p + 16);
  v16h r;
  #pragma unroll
  for (int i = 0; i < 8; ++i) { r[i] = lo[i]; r[8 + i] = hi[i]; }
  return r;
}

// B-fragment: column n = n0 + (lane&15); lanes<16 hold K = k0+{0..7,16..23},
// lanes>=16 hold K = k0+{8..15,24..31}; 2 consecutive K per VGPR.
// `base` is the K-contiguous (transposed) image of B: base[row=n][col=k].
__device__ __forceinline__ v16h load_frag_B(const _Float16* base, int pitch,
                                            int n0, int k0, int lane) {
  int row = n0 + (lane & 15);
  int kb  = k0 + ((lane >> 4) & 1) * 8;
  const _Float16* p = base + row * pitch + kb;
  v8h lo = *(const v8h*)(p);
  v8h hi = *(const v8h*)(p + 16);
  v16h r;
  #pragma unroll
  for (int i = 0; i < 8; ++i) { r[i] = lo[i]; r[8 + i] = hi[i]; }
  return r;
}

// ---- Kernel Z: zero the 128 stat floats ------------------------------------
__global__ void kz(float* stats) {
  if (threadIdx.x < 128) stats[threadIdx.x] = 0.0f;
}

// ---- Kernel A: depthwise k=3 -> pointwise 32x32 -> f16 store + BN1 stats ---
// one 256-thread block per sample; LDS: de tile + weights + stat reduce
#define KA_PITCH 516
__global__ void __launch_bounds__(256) ka(const float* __restrict__ de,
                                          const float* __restrict__ dw_w,
                                          const float* __restrict__ dw_b,
                                          const float* __restrict__ pw_w,
                                          const float* __restrict__ pw_b,
                                          _Float16* __restrict__ xbuf,
                                          float* __restrict__ stats1) {
  extern __shared__ char sm[];
  float* ds_de = (float*)sm;                        // [32][516]
  float* wbuf  = (float*)(sm + CH * KA_PITCH * 4);  // pw_w[1024] pw_b[32] dw_b[32] dw_w[96]
  float* red   = wbuf + 1184;                       // [64]

  const int tid = threadIdx.x;
  const int b   = blockIdx.x;

  for (int i = tid; i < 1024; i += 256) wbuf[i] = pw_w[i];
  if (tid < 32) { wbuf[1024 + tid] = pw_b[tid]; wbuf[1056 + tid] = dw_b[tid]; }
  if (tid < 96) wbuf[1088 + tid] = dw_w[tid];
  if (tid < 64) red[tid] = 0.0f;

  const float* dep = de + (size_t)b * CH * TT;
  for (int i = tid; i < CH * TT / 4; i += 256) {     // 128-byte coalesced b128
    int c = i >> 7, t4 = (i & 127) * 4;
    *(float4*)(ds_de + c * KA_PITCH + t4) = *(const float4*)(dep + c * TT + t4);
  }
  __syncthreads();

  const int t0 = tid * 2;                            // this thread owns t0,t0+1
  float dwa[CH], dwb[CH];
  #pragma unroll
  for (int c = 0; c < CH; ++c) {
    const float* r = ds_de + c * KA_PITCH;
    float w0 = wbuf[1088 + c * 3], w1 = wbuf[1088 + c * 3 + 1],
          w2 = wbuf[1088 + c * 3 + 2], bb = wbuf[1056 + c];
    float xm1 = (t0 == 0) ? 0.0f : r[t0 - 1];
    float x0 = r[t0], x1 = r[t0 + 1];
    float x2 = (t0 + 2 > TT - 1) ? 0.0f : r[t0 + 2];
    dwa[c] = w0 * xm1 + w1 * x0 + w2 * x1 + bb;      // correlation (no flip)
    dwb[c] = w0 * x0  + w1 * x1 + w2 * x2 + bb;
  }

  _Float16* xp = xbuf + (size_t)b * CH * TT;
  for (int o = 0; o < CH; ++o) {
    float y0 = wbuf[1024 + o], y1 = y0;
    #pragma unroll
    for (int c = 0; c < CH; ++c) {
      float w = wbuf[o * CH + c];
      y0 = fmaf(w, dwa[c], y0);
      y1 = fmaf(w, dwb[c], y1);
    }
    float s = y0 + y1, q = y0 * y0 + y1 * y1;        // wave32 tree reduce
    #pragma unroll
    for (int d = 16; d; d >>= 1) { s += __shfl_xor(s, d, 32); q += __shfl_xor(q, d, 32); }
    if ((tid & 31) == 0) { atomicAdd(&red[o], s); atomicAdd(&red[32 + o], q); }
    v2h pk; pk[0] = (_Float16)y0; pk[1] = (_Float16)y1;
    *(v2h*)(xp + o * TT + t0) = pk;                  // coalesced 4B/lane
  }
  __syncthreads();
  if (tid < 64) atomicAdd(&stats1[tid], red[tid]);
}

// ---- Kernel C: fused BN1+relu -> corr -> graphconv -> fc62 partial ---------
// LDS layout (bytes): pitches padded so 16-lane b128 fragment loads hit
// distinct banks and every v8h load is 16B aligned.
#define PX   520   // pitch of xs / g / gc_wT rows (halves)
#define PXT  40    // pitch of xT / cor rows (halves)
#define OFF_XS   0
#define OFF_XT   (OFF_XS + 32 * PX * 2)        // 33280
#define OFF_GW   (OFF_XT + 512 * PXT * 2)      // 74240
#define OFF_COR  (OFF_GW + 16 * PX * 2)        // 90880
#define OFF_S    (OFF_COR + 32 * PXT * 2)      // 93440  f32 [32][33]
#define OFF_GO   (OFF_S + 32 * 33 * 4)         // 97664  f32 [32][17]
#define OFF_BN   (OFF_GO + 32 * 17 * 4)        // 99840  192 f32
#define LDS_KC   (OFF_BN + 192 * 4)            // 100608 bytes (< 320KB WGP)

__global__ void __launch_bounds__(256) kc(const _Float16* __restrict__ xbuf,
                                          const float* __restrict__ stats1,
                                          const float* __restrict__ gc_w,
                                          const float* __restrict__ fc_w,
                                          const float* __restrict__ fc_b,
                                          const float* __restrict__ bn1_g,
                                          const float* __restrict__ bn1_b,
                                          float* __restrict__ hbuf,
                                          float* __restrict__ stats2) {
  extern __shared__ char sm[];
  _Float16* xs = (_Float16*)(sm + OFF_XS);   // [32][PX]  x, later g (overlay)
  _Float16* xT = (_Float16*)(sm + OFF_XT);   // [512][PXT] x transposed
  _Float16* gw = (_Float16*)(sm + OFF_GW);   // [16][PX]  gc_w transposed
  _Float16* cr = (_Float16*)(sm + OFF_COR);  // [32][PXT] correlation f16
  float* S     = (float*)(sm + OFF_S);       // [32][33]  X*X^T
  float* go    = (float*)(sm + OFF_GO);      // [32][17]
  float* bn_sc = (float*)(sm + OFF_BN);
  float* bn_sh = bn_sc + 32;
  float* mu    = bn_sc + 64;
  float* inv   = bn_sc + 96;
  float* msum  = bn_sc + 128;
  float* hred  = bn_sc + 160;

  const int tid  = threadIdx.x;
  const int b    = blockIdx.x;
  const int lane = tid & 31;
  const int wave = tid >> 5;

  // BN1 scale/shift from global stats (finalized by kernel A)
  if (tid < 32) {
    const float cnt = (float)BATCH * (float)TT;
    float m = stats1[tid] / cnt;
    float v = stats1[32 + tid] / cnt - m * m;
    float sc = bn1_g[tid] * rsqrtf(v + EPS);
    bn_sc[tid] = sc;
    bn_sh[tid] = bn1_b[tid] - m * sc;
    hred[tid] = 0.0f;
  }
  for (int i = tid; i < TT * NOUT; i += 256) {       // gc_w -> [16][512] f16
    int t = i >> 4, n = i & 15;
    gw[n * PX + t] = (_Float16)gc_w[i];
  }
  __syncthreads();

  // x = relu(BN(x_pre)); build row-major xs and transposed xT
  const _Float16* xp = xbuf + (size_t)b * CH * TT;
  for (int i2 = tid; i2 < CH * TT / 2; i2 += 256) {  // 4B/lane coalesced
    int c = i2 >> 8, t = (i2 & 255) * 2;
    v2h pr = *(const v2h*)(xp + c * TT + t);
    float a0 = fmaxf((float)pr[0] * bn_sc[c] + bn_sh[c], 0.0f);
    float a1 = fmaxf((float)pr[1] * bn_sc[c] + bn_sh[c], 0.0f);
    _Float16 h0 = (_Float16)a0, h1 = (_Float16)a1;
    v2h w; w[0] = h0; w[1] = h1;
    *(v2h*)(xs + c * PX + t) = w;
    xT[t * PXT + c] = h0;
    xT[(t + 1) * PXT + c] = h1;
  }
  __syncthreads();

  // GEMM1: S = X * X^T  (32x32, K=512) -- waves 0..3, one 16x16 tile each
  if (wave < 4) {
    int m0 = (wave >> 1) * 16, n0 = (wave & 1) * 16;
    v8f acc = {0.f,0.f,0.f,0.f,0.f,0.f,0.f,0.f};
    for (int k0 = 0; k0 < TT; k0 += 32) {
      v16h A = load_frag_A(xs, PX, m0, k0, lane);
      v16h B = load_frag_B(xs, PX, n0, k0, lane);   // B=X^T is K-contig in xs rows
      acc = wmma16(A, B, acc);
    }
    int r0 = m0 + 8 * ((lane >> 4) & 1);
    int c0 = n0 + (lane & 15);
    #pragma unroll
    for (int j = 0; j < 8; ++j) S[(r0 + j) * 33 + c0] = acc[j];
  }
  __syncthreads();

  // per-channel mean (8 threads/channel), then unbiased var from diag(S)
  {
    int c = tid >> 3, j = tid & 7;
    const _Float16* rowp = xs + c * PX;
    float s = 0.0f;
    for (int t = j; t < TT; t += 8) s += (float)rowp[t];
    s += __shfl_xor(s, 1, 32); s += __shfl_xor(s, 2, 32); s += __shfl_xor(s, 4, 32);
    if (j == 0) msum[c] = s;
  }
  __syncthreads();
  if (tid < 32) {
    float m = msum[tid] * (1.0f / TT);
    float vu = (S[tid * 33 + tid] - (float)TT * m * m) * (1.0f / (TT - 1));
    mu[tid] = m;
    inv[tid] = rsqrtf(fmaxf(vu, 1e-30f));
  }
  __syncthreads();
  // cor = (S/T - mu mu^T) * inv_n * inv_m * T/(T-1), stored f16
  {
    const float fT = (float)TT / (float)(TT - 1);
    for (int i = tid; i < CH * CH; i += 256) {
      int n = i >> 5, m = i & 31;
      float cv = (S[n * 33 + m] * (1.0f / TT) - mu[n] * mu[m]) * inv[n] * inv[m] * fT;
      cr[n * PXT + m] = (_Float16)cv;
    }
  }
  __syncthreads();

  // GEMM2: g = cor(32x32) @ x(32x512); K=32 == one WMMA step. 64 tiles / 8 waves.
  // Output overlays xs (x is dead after the mean pass above).
  for (int tile = wave; tile < 64; tile += 8) {
    int m0 = (tile >> 5) * 16, n0 = (tile & 31) * 16;
    v16h A = load_frag_A(cr, PXT, m0, 0, lane);
    v16h B = load_frag_B(xT, PXT, n0, 0, lane);
    v8f acc = {0.f,0.f,0.f,0.f,0.f,0.f,0.f,0.f};
    acc = wmma16(A, B, acc);
    int r0 = m0 + 8 * ((lane >> 4) & 1);
    int c0 = n0 + (lane & 15);
    #pragma unroll
    for (int j = 0; j < 8; ++j) xs[(r0 + j) * PX + c0] = (_Float16)acc[j];
  }
  __syncthreads();

  // GEMM3: go = relu(g(32x512) @ gc_w(512x16)) -- waves 0,1 take the 2 tiles
  if (wave < 2) {
    int m0 = wave * 16;
    v8f acc = {0.f,0.f,0.f,0.f,0.f,0.f,0.f,0.f};
    for (int k0 = 0; k0 < TT; k0 += 32) {
      v16h A = load_frag_A(xs, PX, m0, k0, lane);   // xs now holds g
      v16h B = load_frag_B(gw, PX, 0, k0, lane);
      acc = wmma16(A, B, acc);
    }
    int r0 = m0 + 8 * ((lane >> 4) & 1);
    int c0 = lane & 15;
    #pragma unroll
    for (int j = 0; j < 8; ++j) go[(r0 + j) * 17 + c0] = fmaxf(acc[j], 0.0f);
  }
  __syncthreads();

  // h[n] = sum_i go_flat[i] * fc_w[i][n] + fc_b[n]; 8 partial groups per n
  {
    int n = tid & 31, j = tid >> 5;
    float part = 0.0f;
    for (int i = j * 64; i < j * 64 + 64; ++i) {
      float gv = go[(i >> 4) * 17 + (i & 15)];
      part = fmaf(gv, fc_w[i * CH + n], part);       // lanes: n-contig, coalesced
    }
    atomicAdd(&hred[n], part);
  }
  __syncthreads();
  if (tid < 32) {
    float hv = hred[tid] + fc_b[tid];
    hbuf[(size_t)b * CH + tid] = hv;
    atomicAdd(&stats2[tid], hv);
    atomicAdd(&stats2[32 + tid], hv * hv);
  }
}

// ---- Kernel E: BN2 + sigmoid + classifier ----------------------------------
__global__ void __launch_bounds__(256) ke(const float* __restrict__ hbuf,
                                          const float* __restrict__ stats2,
                                          const float* __restrict__ bn2_g,
                                          const float* __restrict__ bn2_b,
                                          const float* __restrict__ cls_w,
                                          const float* __restrict__ cls_b,
                                          float* __restrict__ out) {
  int idx = blockIdx.x * 256 + threadIdx.x;
  if (idx >= BATCH * NCLS) return;
  int b = idx >> 2, k = idx & 3;
  const float* hb = hbuf + (size_t)b * CH;
  float acc = cls_b[k];
  #pragma unroll
  for (int n = 0; n < CH; ++n) {
    float m = stats2[n] * (1.0f / BATCH);
    float v = stats2[32 + n] * (1.0f / BATCH) - m * m;
    float sc = bn2_g[n] * rsqrtf(v + EPS);
    float hv = (hb[n] - m) * sc + bn2_b[n];
    float s = 1.0f / (1.0f + __expf(-hv));
    acc = fmaf(s, cls_w[n * NCLS + k], acc);
  }
  out[idx] = acc;
}

// ---------------------------------------------------------------------------
extern "C" void kernel_launch(void* const* d_in, const int* in_sizes, int n_in,
                              void* d_out, int out_size, void* d_ws, size_t ws_size,
                              hipStream_t stream) {
  const float* de    = (const float*)d_in[0];
  // d_in[1] = adj (unused by the forward pass)
  const float* dw_w  = (const float*)d_in[2];
  const float* dw_b  = (const float*)d_in[3];
  const float* pw_w  = (const float*)d_in[4];
  const float* pw_b  = (const float*)d_in[5];
  const float* bn1_g = (const float*)d_in[6];
  const float* bn1_b = (const float*)d_in[7];
  const float* gc_w  = (const float*)d_in[8];
  const float* fc_w  = (const float*)d_in[9];
  const float* fc_b  = (const float*)d_in[10];
  const float* bn2_g = (const float*)d_in[11];
  const float* bn2_b = (const float*)d_in[12];
  const float* cls_w = (const float*)d_in[13];
  const float* cls_b = (const float*)d_in[14];

  // workspace: [stats1:64][stats2:64][hbuf:4096*32] f32, then xbuf f16
  float* wsf      = (float*)d_ws;
  float* stats1   = wsf;
  float* stats2   = wsf + 64;
  float* hbuf     = wsf + 128;
  _Float16* xbuf  = (_Float16*)(wsf + 128 + BATCH * CH);   // 134MB f16

  kz<<<1, 128, 0, stream>>>(wsf);
  ka<<<BATCH, 256, (CH * KA_PITCH + 1184 + 64) * 4, stream>>>(
      de, dw_w, dw_b, pw_w, pw_b, xbuf, stats1);
  kc<<<BATCH, 256, LDS_KC, stream>>>(
      xbuf, stats1, gc_w, fc_w, fc_b, bn1_g, bn1_b, hbuf, stats2);
  ke<<<(BATCH * NCLS + 255) / 256, 256, 0, stream>>>(
      hbuf, stats2, bn2_g, bn2_b, cls_w, cls_b, (float*)d_out);
}
